// MixtureOfExperts_24232205484317
// MI455X (gfx1250) — matrix operand order repaired and checked by reading, not compiled, not run on hardware
//
#include <hip/hip_runtime.h>
#include <hip/hip_bf16.h>

// ---------------------------------------------------------------------------
// MoE (B=4,T=1024,D=1024,E=8,H=2048): gate/argmax -> token binning ->
// grouped bf16 WMMA GEMMs (gate/up fused with SiLU) -> down projection.
// Only the selected expert is computed (reference computes all 8, discards 7).
// Block tiles are M-major: the 8 waves of a block share identical B tiles
// (same N columns, different M rows) so the WGP cache serves 7/8 of B traffic.
// ---------------------------------------------------------------------------

#define BT   4096   // B*T tokens
#define DIM  1024   // D
#define EXP  8      // E
#define HID  2048   // H

typedef __attribute__((ext_vector_type(16))) __bf16 v16bf;
typedef __attribute__((ext_vector_type(8)))  float  v8f;

union V16 { v16bf v; uint4 q[2]; };

__device__ __forceinline__ unsigned short f2bf(float f) {
    unsigned int u = __float_as_uint(f);
    unsigned int r = u + 0x7FFFu + ((u >> 16) & 1u);   // round-to-nearest-even
    return (unsigned short)(r >> 16);
}

__device__ __forceinline__ v8f wmma_bf16(const V16& a, const V16& b, v8f c) {
    return __builtin_amdgcn_wmma_f32_16x16x32_bf16(
        /*neg_a=*/false, a.v, /*neg_b=*/false, b.v,
        /*c_mod=*/(short)0, c, /*reuse_a=*/false, /*reuse_b=*/false);
}

// ---------------------------------------------------------------------------
// Zero per-expert counters.
// ---------------------------------------------------------------------------
__global__ __launch_bounds__(32) void init_kernel(int* counts) {
    if (threadIdx.x < EXP) counts[threadIdx.x] = 0;
}

// ---------------------------------------------------------------------------
// Convert an fp32 (E,K,N) weight bank into bf16 tiles laid out as this file's
// WMMA B-operand (32x16 K-by-N tile, contiguous k order as A): lane L holds column (L&15), K-half
// (L>>4)*16, its 16 bf16 elements stored contiguously (32B per lane, 1KB/tile).
// One wave per tile.
// ---------------------------------------------------------------------------
__global__ __launch_bounds__(256) void convert_kernel(const float* __restrict__ src,
                                                      unsigned short* __restrict__ dst,
                                                      int ktiles, int ntiles) {
    int wid  = (blockIdx.x * blockDim.x + threadIdx.x) >> 5;  // global wave = tile id
    int lane = threadIdx.x & 31;
    int tilesPerE = ktiles * ntiles;
    int e  = wid / tilesPerE;
    int r  = wid - e * tilesPerE;
    int kt = r / ntiles;
    int nt = r - kt * ntiles;
    int N = ntiles * 16, K = ktiles * 32;

    const float* s = src + ((size_t)e * K + (size_t)kt * 32) * N + nt * 16 + (lane & 15);
    int khalf = (lane >> 4) * 16;

    unsigned int pk[8];
#pragma unroll
    for (int i = 0; i < 8; ++i) {
        float f0 = s[(size_t)(khalf + 2 * i    ) * N];
        float f1 = s[(size_t)(khalf + 2 * i + 1) * N];
        pk[i] = (unsigned int)f2bf(f0) | ((unsigned int)f2bf(f1) << 16);
    }
    uint4* d4 = (uint4*)(dst + (size_t)wid * 512 + lane * 16);
    d4[0] = make_uint4(pk[0], pk[1], pk[2], pk[3]);
    d4[1] = make_uint4(pk[4], pk[5], pk[6], pk[7]);
}

// ---------------------------------------------------------------------------
// Gating: one block (8 waves) per token. Wave w computes logit for expert w,
// thread 0 does first-max argmax (matches jnp.argmax) and bins the token.
// Also emits the bf16 copy of the token row.
// ---------------------------------------------------------------------------
__global__ __launch_bounds__(256) void gate_kernel(const float* __restrict__ x,
                                                   const float* __restrict__ gate_w,
                                                   int* __restrict__ counts,
                                                   int* __restrict__ bucket,
                                                   unsigned short* __restrict__ xbf) {
    __shared__ float sx[DIM];
    __shared__ float slog[EXP];
    int t   = blockIdx.x;
    int tid = threadIdx.x;
    const float* row = x + (size_t)t * DIM;
#pragma unroll
    for (int i = tid; i < DIM; i += 256) {
        float f = row[i];
        sx[i] = f;
        xbf[(size_t)t * DIM + i] = f2bf(f);
    }
    __syncthreads();

    int w = tid >> 5, lane = tid & 31;
    const float* gw = gate_w + (size_t)w * DIM;
    float acc = 0.f;
    for (int i = lane; i < DIM; i += 32) acc += sx[i] * gw[i];
#pragma unroll
    for (int off = 16; off > 0; off >>= 1) acc += __shfl_down(acc, off);
    if (lane == 0) slog[w] = acc;
    __syncthreads();

    if (tid == 0) {
        int best = 0; float bv = slog[0];
#pragma unroll
        for (int e = 1; e < EXP; ++e) if (slog[e] > bv) { bv = slog[e]; best = e; }
        int pos = atomicAdd(&counts[best], 1);
        bucket[best * BT + pos] = t;
    }
}

// ---------------------------------------------------------------------------
// FFN stage 1: h = silu(x@Wg) * (x@Wu), bf16 out.
// Block tile: M=256 tokens (8 waves x 32) x N=32 cols (gate AND up).
// All waves load the SAME 4 B tiles per K-step (WGP$-served), 2 A tiles each
// => 8 WMMAs per K-step per wave. K loop software-pipelined in registers.
// grid: blockIdx = ((e*16 + mblk)*64 + nblk)
// ---------------------------------------------------------------------------
__global__ __launch_bounds__(256) void ffn1_kernel(const unsigned short* __restrict__ xbf,
                                                   const unsigned short* __restrict__ gbf,
                                                   const unsigned short* __restrict__ ubf,
                                                   const int* __restrict__ counts,
                                                   const int* __restrict__ bucket,
                                                   unsigned short* __restrict__ hbuf) {
    int nb = blockIdx.x & 63;            // 64 n-blocks of 32 cols
    int mb = (blockIdx.x >> 6) & 15;     // 16 m-blocks of 256 tokens
    int e  = blockIdx.x >> 10;
    int count = counts[e];
    if (mb * 256 >= count) return;       // uniform early-exit (EXEC stays full)

    int w = threadIdx.x >> 5, lane = threadIdx.x & 31;
    int mbase = mb * 256 + w * 32;       // this wave's 32 rows
    int n0 = nb * 32;
    int nt0 = n0 >> 4;

    // Two A-operand rows per lane (ISA 16-bit A layout: lane&15 = row in tile)
    int m = lane & 15;
    int r0 = mbase + m;      if (r0 >= count) r0 = count - 1;
    int r1 = mbase + 16 + m; if (r1 >= count) r1 = count - 1;
    int tok0 = bucket[e * BT + r0];
    int tok1 = bucket[e * BT + r1];
    const unsigned short* arow0 = xbf + (size_t)tok0 * DIM;
    const unsigned short* arow1 = xbf + (size_t)tok1 * DIM;
    int khalf8 = (lane >> 4) * 16;

    const size_t kstep = (size_t)128 * 512;                      // ushorts between K-tiles
    const unsigned short* gB = gbf + (((size_t)e * 32) * 128 + nt0) * 512 + lane * 16;
    const unsigned short* uB = ubf + (((size_t)e * 32) * 128 + nt0) * 512 + lane * 16;

    auto loadA = [&](V16* Ad, int kt) {
        const unsigned short* p0 = arow0 + kt * 32 + khalf8;
        Ad[0].q[0] = *(const uint4*)(p0);
        Ad[0].q[1] = *(const uint4*)(p0 + 8);
        const unsigned short* p1 = arow1 + kt * 32 + khalf8;
        Ad[1].q[0] = *(const uint4*)(p1);
        Ad[1].q[1] = *(const uint4*)(p1 + 8);
    };
    auto loadB = [&](V16* Bd, int kt) {
        const uint4* g0 = (const uint4*)(gB + kt * kstep);
        const uint4* g1 = (const uint4*)(gB + kt * kstep + 512);
        const uint4* u0 = (const uint4*)(uB + kt * kstep);
        const uint4* u1 = (const uint4*)(uB + kt * kstep + 512);
        Bd[0].q[0] = g0[0]; Bd[0].q[1] = g0[1];
        Bd[1].q[0] = g1[0]; Bd[1].q[1] = g1[1];
        Bd[2].q[0] = u0[0]; Bd[2].q[1] = u0[1];
        Bd[3].q[0] = u1[0]; Bd[3].q[1] = u1[1];
    };

    v8f accG[2][2] = {{{}, {}}, {{}, {}}};   // [mtile][ntile]
    v8f accU[2][2] = {{{}, {}}, {{}, {}}};

    V16 A[2], B[4], An[2], Bn[4];
    loadA(A, 0);
    loadB(B, 0);

    for (int kt = 0; kt < DIM / 32; ++kt) {
        if (kt + 1 < DIM / 32) {             // issue next-step loads first
            if (w == 0) {
                __builtin_prefetch(gB + (kt + 6) * kstep, 0, 3);
                __builtin_prefetch(uB + (kt + 6) * kstep, 0, 3);
            }
            loadA(An, kt + 1);
            loadB(Bn, kt + 1);
        }
#pragma unroll
        for (int i = 0; i < 2; ++i) {
            accG[i][0] = wmma_bf16(A[i], B[0], accG[i][0]);
            accG[i][1] = wmma_bf16(A[i], B[1], accG[i][1]);
            accU[i][0] = wmma_bf16(A[i], B[2], accU[i][0]);
            accU[i][1] = wmma_bf16(A[i], B[3], accU[i][1]);
        }
        A[0] = An[0]; A[1] = An[1];
        B[0] = Bn[0]; B[1] = Bn[1]; B[2] = Bn[2]; B[3] = Bn[3];
    }

    // Epilogue: h = silu(a)*u, scatter bf16 rows via bucket.
#pragma unroll
    for (int i = 0; i < 2; ++i) {
#pragma unroll
        for (int j = 0; j < 2; ++j) {
            int col = n0 + j * 16 + (lane & 15);
#pragma unroll
            for (int r = 0; r < 8; ++r) {
                int ri = mbase + i * 16 + (lane >> 4) * 8 + r;     // C layout row
                if (ri < count) {
                    float av = accG[i][j][r];
                    float hv = (av / (1.f + __expf(-av))) * accU[i][j][r];
                    int tk = bucket[e * BT + ri];
                    hbuf[(size_t)tk * HID + col] = f2bf(hv);
                }
            }
        }
    }
}

// ---------------------------------------------------------------------------
// FFN stage 2: y = h @ Wd[e], fp32 out.
// Block tile: M=256 x N=64; wave tile M=32 x N=64 (2 A x 4 shared B tiles)
// => 8 WMMAs per K-step, pipelined as above.
// grid: blockIdx = ((e*16 + mblk)*16 + nblk)
// ---------------------------------------------------------------------------
__global__ __launch_bounds__(256) void ffn2_kernel(const unsigned short* __restrict__ hbuf,
                                                   const unsigned short* __restrict__ dbf,
                                                   const int* __restrict__ counts,
                                                   const int* __restrict__ bucket,
                                                   float* __restrict__ out) {
    int nb = blockIdx.x & 15;            // 16 n-blocks of 64 cols
    int mb = (blockIdx.x >> 4) & 15;     // 16 m-blocks of 256 tokens
    int e  = blockIdx.x >> 8;
    int count = counts[e];
    if (mb * 256 >= count) return;

    int w = threadIdx.x >> 5, lane = threadIdx.x & 31;
    int mbase = mb * 256 + w * 32;
    int n0 = nb * 64;
    int nt0 = n0 >> 4;

    int m = lane & 15;
    int r0 = mbase + m;      if (r0 >= count) r0 = count - 1;
    int r1 = mbase + 16 + m; if (r1 >= count) r1 = count - 1;
    int tok0 = bucket[e * BT + r0];
    int tok1 = bucket[e * BT + r1];
    const unsigned short* arow0 = hbuf + (size_t)tok0 * HID;
    const unsigned short* arow1 = hbuf + (size_t)tok1 * HID;
    int khalf8 = (lane >> 4) * 16;

    const size_t kstep = (size_t)64 * 512;
    const unsigned short* dB = dbf + (((size_t)e * 64) * 64 + nt0) * 512 + lane * 16;

    auto loadA = [&](V16* Ad, int kt) {
        const unsigned short* p0 = arow0 + kt * 32 + khalf8;
        Ad[0].q[0] = *(const uint4*)(p0);
        Ad[0].q[1] = *(const uint4*)(p0 + 8);
        const unsigned short* p1 = arow1 + kt * 32 + khalf8;
        Ad[1].q[0] = *(const uint4*)(p1);
        Ad[1].q[1] = *(const uint4*)(p1 + 8);
    };
    auto loadB = [&](V16* Bd, int kt) {
#pragma unroll
        for (int j = 0; j < 4; ++j) {
            const uint4* p = (const uint4*)(dB + kt * kstep + j * 512);
            Bd[j].q[0] = p[0]; Bd[j].q[1] = p[1];
        }
    };

    v8f acc[2][4] = {{{}, {}, {}, {}}, {{}, {}, {}, {}}};

    V16 A[2], B[4], An[2], Bn[4];
    loadA(A, 0);
    loadB(B, 0);

    for (int kt = 0; kt < HID / 32; ++kt) {
        if (kt + 1 < HID / 32) {
            if (w == 0) __builtin_prefetch(dB + (kt + 6) * kstep, 0, 3);
            loadA(An, kt + 1);
            loadB(Bn, kt + 1);
        }
#pragma unroll
        for (int i = 0; i < 2; ++i) {
#pragma unroll
            for (int j = 0; j < 4; ++j) {
                acc[i][j] = wmma_bf16(A[i], B[j], acc[i][j]);
            }
        }
        A[0] = An[0]; A[1] = An[1];
        B[0] = Bn[0]; B[1] = Bn[1]; B[2] = Bn[2]; B[3] = Bn[3];
    }

#pragma unroll
    for (int i = 0; i < 2; ++i) {
#pragma unroll
        for (int j = 0; j < 4; ++j) {
            int col = n0 + j * 16 + (lane & 15);
#pragma unroll
            for (int r = 0; r < 8; ++r) {
                int ri = mbase + i * 16 + (lane >> 4) * 8 + r;
                if (ri < count) {
                    int tk = bucket[e * BT + ri];
                    out[(size_t)tk * DIM + col] = acc[i][j][r];
                }
            }
        }
    }
}

// ---------------------------------------------------------------------------
// Host launcher. Workspace layout (bytes):
//   [0, 32)        counts[8]
//   [1K, 129K)     bucket[E*BT]
//   [1M, 9M)       x bf16            (BT*D*2)
//   [16M, 32M)     h bf16            (BT*H*2)
//   [32M, 64M)     gate_bank bf16 tiled
//   [64M, 96M)     up_bank   bf16 tiled
//   [96M, 128M)    down_bank bf16 tiled
// ---------------------------------------------------------------------------
extern "C" void kernel_launch(void* const* d_in, const int* in_sizes, int n_in,
                              void* d_out, int out_size, void* d_ws, size_t ws_size,
                              hipStream_t stream) {
    const float* x         = (const float*)d_in[0];
    const float* gate_w    = (const float*)d_in[1];
    const float* gate_bank = (const float*)d_in[2];
    const float* up_bank   = (const float*)d_in[3];
    const float* down_bank = (const float*)d_in[4];
    float* out = (float*)d_out;

    char* ws = (char*)d_ws;
    int* counts  = (int*)(ws + 0);
    int* bucket  = (int*)(ws + 1024);
    unsigned short* xbf  = (unsigned short*)(ws + ((size_t)1  << 20));
    unsigned short* hbuf = (unsigned short*)(ws + ((size_t)16 << 20));
    unsigned short* gbf  = (unsigned short*)(ws + ((size_t)32 << 20));
    unsigned short* ubf  = (unsigned short*)(ws + ((size_t)64 << 20));
    unsigned short* dbf  = (unsigned short*)(ws + ((size_t)96 << 20));

    init_kernel<<<1, 32, 0, stream>>>(counts);

    // 8 waves/block, one wave per 32x16 tile. E*ktiles*ntiles = 32768 tiles each.
    convert_kernel<<<4096, 256, 0, stream>>>(gate_bank, gbf, DIM / 32, HID / 16);
    convert_kernel<<<4096, 256, 0, stream>>>(up_bank,   ubf, DIM / 32, HID / 16);
    convert_kernel<<<4096, 256, 0, stream>>>(down_bank, dbf, HID / 32, DIM / 16);

    gate_kernel<<<BT, 256, 0, stream>>>(x, gate_w, counts, bucket, xbf);

    // E * 16 m-blocks(256 tokens) * 64 n-blocks; empty blocks exit on counts[e].
    ffn1_kernel<<<EXP * 16 * 64, 256, 0, stream>>>(xbf, gbf, ubf, counts, bucket, hbuf);
    // E * 16 m-blocks * 16 n-blocks (64 cols per block)
    ffn2_kernel<<<EXP * 16 * 16, 256, 0, stream>>>(hbuf, dbf, counts, bucket, out);
}
